// MultiHeadSelfAttention_20160576487445
// MI455X (gfx1250) — compile-verified, matrix operand order checked by reading, not run on hardware
//
#include <hip/hip_runtime.h>

#define D_MODEL 1024
#define N_HEADS 16
#define HEAD_DIM 64
#define BATCH 2
#define SEQ 2048
#define M_TOTAL (BATCH * SEQ) /* 4096 */

typedef __attribute__((ext_vector_type(16))) _Float16 v16h;
typedef __attribute__((ext_vector_type(8)))  _Float16 v8h;
typedef __attribute__((ext_vector_type(8)))  float    v8f;

// Load one lane's 16-half A/B fragment: two contiguous 16-byte chunks at p and p+16
// (CDNA5 16-bit fragment layout: K = base8+0..7 then 16+base8+0..7 per lane).
__device__ __forceinline__ v16h load_frag(const _Float16* p) {
  v8h lo = *(const v8h*)(p);
  v8h hi = *(const v8h*)(p + 16);
  v16h r;
#pragma unroll
  for (int i = 0; i < 8; ++i) { r[i] = lo[i]; r[i + 8] = hi[i]; }
  return r;
}

__device__ __forceinline__ v8f wmma_f16(v16h a, v16h b, v8f c) {
  return __builtin_amdgcn_wmma_f32_16x16x32_f16(false, a, false, b, (short)0, c,
                                                false, false);
}

__global__ void __launch_bounds__(256)
cvt_f32_to_f16(const float* __restrict__ s, _Float16* __restrict__ d, int n) {
  int i = blockIdx.x * blockDim.x + threadIdx.x;
  int st = gridDim.x * blockDim.x;
  for (; i < n; i += st) d[i] = (_Float16)s[i];
}

// Y[m,n] = sum_k Xh[m,k] * Wh[n,k] + bias[n]   (torch Linear semantics)
// mode 0: store f16 into out[(b*16+h)*SEQ*64 + s*64 + d]   (Q/K head-major)
// mode 1: store f16 into out[(b*16+h)*64*SEQ + d*SEQ + s]  (V transposed)
__global__ void __launch_bounds__(256)
proj_gemm_f16(const _Float16* __restrict__ Xh, const _Float16* __restrict__ Wh,
              const float* __restrict__ bias, _Float16* __restrict__ out, int mode) {
  const int wid = threadIdx.x >> 5;
  const int w = blockIdx.x * 8 + wid;      // 4096 wave jobs
  const int m0 = (w >> 4) * 16;            // 256 row tiles
  const int n0 = (w & 15) * 64;            // 16 col strips of 64
  const int lane = threadIdx.x & 31;
  const int l16 = lane & 15;
  const int half = (lane & 16) ? 8 : 0;

  const _Float16* aptr = Xh + (size_t)(m0 + l16) * D_MODEL + half;
  const _Float16* bp0 = Wh + (size_t)(n0 + 0 * 16 + l16) * D_MODEL + half;
  const _Float16* bp1 = Wh + (size_t)(n0 + 1 * 16 + l16) * D_MODEL + half;
  const _Float16* bp2 = Wh + (size_t)(n0 + 2 * 16 + l16) * D_MODEL + half;
  const _Float16* bp3 = Wh + (size_t)(n0 + 3 * 16 + l16) * D_MODEL + half;

  v8f acc[4] = {};
  // software pipeline: fragments for step k live while step k+1 loads are in flight
  v16h a  = load_frag(aptr);
  v16h b0 = load_frag(bp0);
  v16h b1 = load_frag(bp1);
  v16h b2 = load_frag(bp2);
  v16h b3 = load_frag(bp3);
  for (int k0 = 32; k0 < D_MODEL; k0 += 32) {
    v16h an  = load_frag(aptr + k0);
    v16h bn0 = load_frag(bp0 + k0);
    v16h bn1 = load_frag(bp1 + k0);
    v16h bn2 = load_frag(bp2 + k0);
    v16h bn3 = load_frag(bp3 + k0);
    acc[0] = wmma_f16(a, b0, acc[0]);
    acc[1] = wmma_f16(a, b1, acc[1]);
    acc[2] = wmma_f16(a, b2, acc[2]);
    acc[3] = wmma_f16(a, b3, acc[3]);
    a = an; b0 = bn0; b1 = bn1; b2 = bn2; b3 = bn3;
  }
  acc[0] = wmma_f16(a, b0, acc[0]);
  acc[1] = wmma_f16(a, b1, acc[1]);
  acc[2] = wmma_f16(a, b2, acc[2]);
  acc[3] = wmma_f16(a, b3, acc[3]);

#pragma unroll
  for (int t = 0; t < 4; ++t) {
    const int n = n0 + t * 16 + l16;
    const float bv = bias[n];
    const int h = n >> 6, d = n & (HEAD_DIM - 1);
#pragma unroll
    for (int r = 0; r < 8; ++r) {
      const int m = m0 + r + half;         // C layout: row = r + 8*(lane>=16)
      const int b = m >> 11, s = m & (SEQ - 1);
      const float val = acc[t][r] + bv;
      if (mode == 0)
        out[((size_t)(b * N_HEADS + h) * SEQ + s) * HEAD_DIM + d] = (_Float16)val;
      else
        out[((size_t)(b * N_HEADS + h) * HEAD_DIM + d) * SEQ + s] = (_Float16)val;
    }
  }
}

// Flash attention: one wave handles 16 query rows of one (batch, head).
__global__ void __launch_bounds__(256)
flash_attn(const _Float16* __restrict__ Qh, const _Float16* __restrict__ Kh,
           const _Float16* __restrict__ Vt, const int* __restrict__ mask,
           _Float16* __restrict__ Ah) {
  __shared__ __align__(16) _Float16 lds[8 * 16 * 32];
  const int wid = threadIdx.x >> 5;
  const int bh = blockIdx.y;               // b*16+h
  const int b = bh >> 4, h = bh & 15;
  const int q0 = (blockIdx.x * 8 + wid) * 16;
  const int lane = threadIdx.x & 31;
  const int l16 = lane & 15;
  const int half = (lane & 16) ? 8 : 0;

  const _Float16* Qb = Qh + (size_t)bh * SEQ * HEAD_DIM;
  const _Float16* Kb = Kh + (size_t)bh * SEQ * HEAD_DIM;
  const _Float16* Vb = Vt + (size_t)bh * SEQ * HEAD_DIM; // [d][s]
  const int* mb = mask + b * SEQ;
  _Float16* L = lds + wid * (16 * 32);

  v16h qf[2];
#pragma unroll
  for (int c = 0; c < 2; ++c)
    qf[c] = load_frag(Qb + (size_t)(q0 + l16) * HEAD_DIM + c * 32 + half);

  v8f o[4] = {};
  float mrow[8], lrow[8];
#pragma unroll
  for (int r = 0; r < 8; ++r) { mrow[r] = -3.0e38f; lrow[r] = 0.0f; }

  for (int kb = 0; kb <= q0 + 15; kb += 32) {
    // Issue ALL loads for this key block up front: K fragments feed the score
    // WMMAs immediately; V fragments are consumed only after softmax, so their
    // latency hides behind the score WMMAs + exp/shuffle VALU + LDS round-trip.
    v16h kf00 = load_frag(Kb + (size_t)(kb + l16) * HEAD_DIM + 0 + half);
    v16h kf01 = load_frag(Kb + (size_t)(kb + l16) * HEAD_DIM + 32 + half);
    v16h kf10 = load_frag(Kb + (size_t)(kb + 16 + l16) * HEAD_DIM + 0 + half);
    v16h kf11 = load_frag(Kb + (size_t)(kb + 16 + l16) * HEAD_DIM + 32 + half);
    v16h vf[4];
#pragma unroll
    for (int t = 0; t < 4; ++t)
      vf[t] = load_frag(Vb + (size_t)(t * 16 + l16) * SEQ + kb + half);

    v8f s0 = {}, s1 = {};
    s0 = wmma_f16(qf[0], kf00, s0);
    s0 = wmma_f16(qf[1], kf01, s0);
    s1 = wmma_f16(qf[0], kf10, s1);
    s1 = wmma_f16(qf[1], kf11, s1);

    const int k0i = kb + l16, k1i = kb + 16 + l16;
    const bool pm0 = (mb[k0i] != 0), pm1 = (mb[k1i] != 0);
#pragma unroll
    for (int r = 0; r < 8; ++r) {
      const int q = q0 + r + half;
      float a0 = (pm0 && k0i <= q) ? s0[r] * 0.125f : -3.0e38f;
      float a1 = (pm1 && k1i <= q) ? s1[r] * 0.125f : -3.0e38f;
      float tmax = fmaxf(a0, a1);
#pragma unroll
      for (int off = 8; off; off >>= 1) tmax = fmaxf(tmax, __shfl_xor(tmax, off));
      const float mnew = fmaxf(mrow[r], tmax);
      const float p0 = __expf(a0 - mnew);
      const float p1 = __expf(a1 - mnew);
      float ps = p0 + p1;
#pragma unroll
      for (int off = 8; off; off >>= 1) ps += __shfl_xor(ps, off);
      const float corr = __expf(mrow[r] - mnew);
      mrow[r] = mnew;
      lrow[r] = lrow[r] * corr + ps;
#pragma unroll
      for (int t = 0; t < 4; ++t) o[t][r] *= corr;
      // stash P (16x32 row-major) for re-layout into A-fragment pattern
      L[(r + half) * 32 + l16] = (_Float16)p0;
      L[(r + half) * 32 + 16 + l16] = (_Float16)p1;
    }
    asm volatile("s_wait_dscnt 0" ::: "memory");
    v16h pf = load_frag(L + l16 * 32 + half);
#pragma unroll
    for (int t = 0; t < 4; ++t) o[t] = wmma_f16(pf, vf[t], o[t]);
  }

  const size_t obase = ((size_t)b * SEQ) * D_MODEL + (size_t)h * HEAD_DIM;
#pragma unroll
  for (int r = 0; r < 8; ++r) {
    const float inv = 1.0f / lrow[r];
    const int s = q0 + r + half;
#pragma unroll
    for (int t = 0; t < 4; ++t)
      Ah[obase + (size_t)s * D_MODEL + t * 16 + l16] = (_Float16)(o[t][r] * inv);
  }
}

// out[m,n] = sum_k Ah[m,k] * Wo[n,k] + bo[n], fp32 output
__global__ void __launch_bounds__(256)
out_proj_gemm(const _Float16* __restrict__ Ah, const _Float16* __restrict__ Woh,
              const float* __restrict__ bo, float* __restrict__ out) {
  const int wid = threadIdx.x >> 5;
  const int w = blockIdx.x * 8 + wid;
  const int m0 = (w >> 4) * 16;
  const int n0 = (w & 15) * 64;
  const int lane = threadIdx.x & 31;
  const int l16 = lane & 15;
  const int half = (lane & 16) ? 8 : 0;

  const _Float16* aptr = Ah + (size_t)(m0 + l16) * D_MODEL + half;
  const _Float16* bp0 = Woh + (size_t)(n0 + 0 * 16 + l16) * D_MODEL + half;
  const _Float16* bp1 = Woh + (size_t)(n0 + 1 * 16 + l16) * D_MODEL + half;
  const _Float16* bp2 = Woh + (size_t)(n0 + 2 * 16 + l16) * D_MODEL + half;
  const _Float16* bp3 = Woh + (size_t)(n0 + 3 * 16 + l16) * D_MODEL + half;

  v8f acc[4] = {};
  v16h a  = load_frag(aptr);
  v16h b0 = load_frag(bp0);
  v16h b1 = load_frag(bp1);
  v16h b2 = load_frag(bp2);
  v16h b3 = load_frag(bp3);
  for (int k0 = 32; k0 < D_MODEL; k0 += 32) {
    v16h an  = load_frag(aptr + k0);
    v16h bn0 = load_frag(bp0 + k0);
    v16h bn1 = load_frag(bp1 + k0);
    v16h bn2 = load_frag(bp2 + k0);
    v16h bn3 = load_frag(bp3 + k0);
    acc[0] = wmma_f16(a, b0, acc[0]);
    acc[1] = wmma_f16(a, b1, acc[1]);
    acc[2] = wmma_f16(a, b2, acc[2]);
    acc[3] = wmma_f16(a, b3, acc[3]);
    a = an; b0 = bn0; b1 = bn1; b2 = bn2; b3 = bn3;
  }
  acc[0] = wmma_f16(a, b0, acc[0]);
  acc[1] = wmma_f16(a, b1, acc[1]);
  acc[2] = wmma_f16(a, b2, acc[2]);
  acc[3] = wmma_f16(a, b3, acc[3]);

#pragma unroll
  for (int t = 0; t < 4; ++t) {
    const int n = n0 + t * 16 + l16;
    const float bv = bo[n];
#pragma unroll
    for (int r = 0; r < 8; ++r)
      out[(size_t)(m0 + r + half) * D_MODEL + n] = acc[t][r] + bv;
  }
}

extern "C" void kernel_launch(void* const* d_in, const int* in_sizes, int n_in,
                              void* d_out, int out_size, void* d_ws, size_t ws_size,
                              hipStream_t stream) {
  const float* x  = (const float*)d_in[0];
  const int* mask = (const int*)d_in[1];
  const float* Wq = (const float*)d_in[2];
  const float* bq = (const float*)d_in[3];
  const float* Wk = (const float*)d_in[4];
  const float* bk = (const float*)d_in[5];
  const float* Wv = (const float*)d_in[6];
  const float* bv = (const float*)d_in[7];
  const float* Wo = (const float*)d_in[8];
  const float* bo = (const float*)d_in[9];

  char* ws = (char*)d_ws;
  size_t off = 0;
  const size_t XB = (size_t)M_TOTAL * D_MODEL * sizeof(_Float16); // 8 MB
  const size_t WB = (size_t)D_MODEL * D_MODEL * sizeof(_Float16); // 2 MB
  _Float16* Xh  = (_Float16*)(ws + off); off += XB;
  _Float16* Wqh = (_Float16*)(ws + off); off += WB;
  _Float16* Wkh = (_Float16*)(ws + off); off += WB;
  _Float16* Wvh = (_Float16*)(ws + off); off += WB;
  _Float16* Woh = (_Float16*)(ws + off); off += WB;
  _Float16* Qh  = (_Float16*)(ws + off); off += XB;
  _Float16* Kh  = (_Float16*)(ws + off); off += XB;
  _Float16* Vt  = (_Float16*)(ws + off); off += XB;
  _Float16* Ah  = (_Float16*)(ws + off); off += XB; // 48 MB total

  cvt_f32_to_f16<<<1024, 256, 0, stream>>>(x, Xh, M_TOTAL * D_MODEL);
  cvt_f32_to_f16<<<512, 256, 0, stream>>>(Wq, Wqh, D_MODEL * D_MODEL);
  cvt_f32_to_f16<<<512, 256, 0, stream>>>(Wk, Wkh, D_MODEL * D_MODEL);
  cvt_f32_to_f16<<<512, 256, 0, stream>>>(Wv, Wvh, D_MODEL * D_MODEL);
  cvt_f32_to_f16<<<512, 256, 0, stream>>>(Wo, Woh, D_MODEL * D_MODEL);

  proj_gemm_f16<<<512, 256, 0, stream>>>(Xh, Wqh, bq, Qh, 0);
  proj_gemm_f16<<<512, 256, 0, stream>>>(Xh, Wkh, bk, Kh, 0);
  proj_gemm_f16<<<512, 256, 0, stream>>>(Xh, Wvh, bv, Vt, 1);

  flash_attn<<<dim3(16, 32), 256, 0, stream>>>(Qh, Kh, Vt, mask, Ah);

  out_proj_gemm<<<512, 256, 0, stream>>>(Ah, Woh, bo, (float*)d_out);
}
